// NNK_10093173146229
// MI455X (gfx1250) — matrix operand order repaired and checked by reading, not compiled, not run on hardware
//
#include <hip/hip_runtime.h>
#include <hip/hip_bf16.h>
#include <math.h>

// Problem constants (from reference)
#define DIM_K    1024
#define NRF      2048     // NUM_RFS (m)
#define OUTD     4096     // OUT_DIM
#define BSROWS   8192     // B*S

// GEMM tiling: block tile 128x128, BK=32, 8 waves as 2(M) x 4(N), wave tile 64x32
#define BM 128
#define BN 128
#define BK 32
#define LDSS 48           // padded LDS row stride in bf16 (96B, multiple of 16B)

typedef __attribute__((ext_vector_type(16))) __bf16 v16bf;
typedef __attribute__((ext_vector_type(8)))  float  v8f;

// ---------------------------------------------------------------------------
// CDNA5 async global->LDS path (ASYNCcnt), with safe fallback
// ---------------------------------------------------------------------------
#if defined(__gfx1250__) && __has_builtin(__builtin_amdgcn_global_load_async_to_lds_b128)
#define USE_ASYNC 1
#else
#define USE_ASYNC 0
#endif

#if USE_ASYNC
typedef int gv4i __attribute__((vector_size(4 * sizeof(int))));
#define AS1 __attribute__((address_space(1)))
#define AS3 __attribute__((address_space(3)))
__device__ __forceinline__ void async_copy_b128(__bf16* lds, const __bf16* g) {
    // emits global_load_async_to_lds_b128 (tracked with ASYNCcnt)
    __builtin_amdgcn_global_load_async_to_lds_b128(
        (AS1 gv4i*)(gv4i*)(void*)(const_cast<__bf16*>(g)),
        (AS3 gv4i*)(gv4i*)(void*)(lds),
        /*imm offset*/0, /*cpol*/0);
}
__device__ __forceinline__ void async_wait0() {
#if __has_builtin(__builtin_amdgcn_s_wait_asynccnt)
    __builtin_amdgcn_s_wait_asynccnt(0);
#else
    asm volatile("s_wait_asynccnt 0x0" ::: "memory");
#endif
}
#endif

// ---------------------------------------------------------------------------
// f32 -> bf16 row convert + per-row sum of squares (sq may be null)
// ---------------------------------------------------------------------------
__global__ __launch_bounds__(256)
void rowconv_kernel(const float* __restrict__ in, __bf16* __restrict__ outb,
                    float* __restrict__ sq, int K)
{
    __shared__ float red[256];
    const int row = blockIdx.x;
    const float* r = in + (size_t)row * K;
    __bf16* o = outb + (size_t)row * K;

    float s = 0.f;
    for (int k = threadIdx.x; k < K; k += 256) {
        float v = r[k];
        s += v * v;
        o[k] = (__bf16)v;
    }
    if (sq != nullptr) {
        red[threadIdx.x] = s;
        __syncthreads();
        for (int off = 128; off > 0; off >>= 1) {
            if (threadIdx.x < off) red[threadIdx.x] += red[threadIdx.x + off];
            __syncthreads();
        }
        if (threadIdx.x == 0) sq[row] = red[0];
    }
}

// ---------------------------------------------------------------------------
// WMMA GEMM:  C[M,N] = A[M,K] * B[N,K]^T   (A, B bf16 row-major, K contiguous)
// EPI == 0 : phi epilogue  out_bf16[row,col] = inv_sqrt_m*exp(ab*d - 0.5*sq*ab^2)
// EPI == 1 : plain f32 store to Out
// ---------------------------------------------------------------------------
template <int EPI>
__global__ __launch_bounds__(256)
void gemm_wmma_kernel(const __bf16* __restrict__ A, const __bf16* __restrict__ Bm,
                      void* __restrict__ Out,
                      const float* __restrict__ sqrow, const float* __restrict__ xis,
                      int M, int N, int K)
{
    __shared__ __bf16 lA[2][BM * LDSS];
    __shared__ __bf16 lB[2][BN * LDSS];
    __shared__ float  sSq[BM];
    __shared__ float  sXi[BN];
    (void)M;

    const int tid  = threadIdx.x;
    const int lane = tid & 31;
    const int wave = tid >> 5;
    const int wmw  = wave >> 2;     // 0..1  (M direction, 64 rows each)
    const int wnw  = wave & 3;      // 0..3  (N direction, 32 cols each)
    const int m0   = blockIdx.y * BM;
    const int n0   = blockIdx.x * BN;

    if constexpr (EPI == 0) {
        if (tid < BM) sSq[tid] = sqrow[m0 + tid];
        if (tid < BN) sXi[tid] = xis[n0 + tid];
    }

    v8f acc[4][2];
#pragma unroll
    for (int i = 0; i < 4; ++i)
#pragma unroll
        for (int j = 0; j < 2; ++j) acc[i][j] = (v8f){};

    // Global->LDS mapping: 16B chunks (8 bf16). Tile = 128 rows x 32 cols ->
    // 512 chunks; thread t handles rows (t>>2) and (t>>2)+64, col (t&3)*8.
    const int row  = tid >> 2;          // 0..63
    const int col8 = (tid & 3) * 8;     // 0,8,16,24
    const int c    = lane & 15;         // position within 16
    const int half = lane >> 4;         // lane group

    const __bf16* gA0 = A  + (size_t)(m0 + row)      * K + col8;
    const __bf16* gA1 = A  + (size_t)(m0 + row + 64) * K + col8;
    const __bf16* gB0 = Bm + (size_t)(n0 + row)      * K + col8;
    const __bf16* gB1 = Bm + (size_t)(n0 + row + 64) * K + col8;

    const int lA0 = (row)      * LDSS + col8;
    const int lA1 = (row + 64) * LDSS + col8;
    const int nT = K / BK;

#if USE_ASYNC
    // ---------------- async double-buffered pipeline ----------------
    async_copy_b128(&lA[0][lA0], gA0);
    async_copy_b128(&lA[0][lA1], gA1);
    async_copy_b128(&lB[0][lA0], gB0);
    async_copy_b128(&lB[0][lA1], gB1);

    int cur = 0;
    for (int kt = 0; kt < nT; ++kt) {
        async_wait0();          // this wave's tile landed in LDS
        __syncthreads();        // everyone's tile landed; prev buffer free
        if (kt + 1 < nT) {
            const int ko = (kt + 1) * BK;
            const int nb = cur ^ 1;
            async_copy_b128(&lA[nb][lA0], gA0 + ko);
            async_copy_b128(&lA[nb][lA1], gA1 + ko);
            async_copy_b128(&lB[nb][lA0], gB0 + ko);
            async_copy_b128(&lB[nb][lA1], gB1 + ko);
        }
#else
    // ---------------- fallback: register staging ----------------
    int cur = 0;
    for (int kt = 0; kt < nT; ++kt) {
        const int ko = kt * BK;
        const uint4 ra0 = *(const uint4*)(gA0 + ko);
        const uint4 ra1 = *(const uint4*)(gA1 + ko);
        const uint4 rb0 = *(const uint4*)(gB0 + ko);
        const uint4 rb1 = *(const uint4*)(gB1 + ko);
        __syncthreads();
        *(uint4*)(&lA[0][lA0]) = ra0;
        *(uint4*)(&lA[0][lA1]) = ra1;
        *(uint4*)(&lB[0][lA0]) = rb0;
        *(uint4*)(&lB[0][lA1]) = rb1;
        __syncthreads();
#endif
        // A fragment: lane (half,c) = row c of 16-row tile,
        //             K = {8*half..8*half+7} U {16+8*half..16+8*half+7}
        union { v16bf v; uint4 u[2]; } fa[4], fb[2];
#pragma unroll
        for (int i = 0; i < 4; ++i) {
            const __bf16* base = &lA[cur][(size_t)(wmw * 64 + i * 16 + c) * LDSS];
            fa[i].u[0] = *(const uint4*)(base + 8 * half);
            fa[i].u[1] = *(const uint4*)(base + 16 + 8 * half);
        }
        // B fragment: lane (half,c) = column c, K = {8*half..8*half+7} U {16+8*half..16+8*half+7}
#pragma unroll
        for (int j = 0; j < 2; ++j) {
            const __bf16* base = &lB[cur][(size_t)(wnw * 32 + j * 16 + c) * LDSS + 8 * half];
            fb[j].u[0] = *(const uint4*)(base);
            fb[j].u[1] = *(const uint4*)(base + 16);
        }

#pragma unroll
        for (int i = 0; i < 4; ++i)
#pragma unroll
            for (int j = 0; j < 2; ++j)
                acc[i][j] = __builtin_amdgcn_wmma_f32_16x16x32_bf16(
                    /*neg_a=*/false, fa[i].v, /*neg_b=*/false, fb[j].v,
                    /*c_mod=*/(short)0, acc[i][j],
                    /*reuse_a=*/false, /*reuse_b=*/false);
#if USE_ASYNC
        cur ^= 1;
#endif
    }

    // Epilogue. C/D layout: lane -> N = c, VGPR e -> M = e + 8*half (per 16x16 tile)
    if constexpr (EPI == 0) {
        __bf16* out = (__bf16*)Out;
        const float inv_sqrt_m = 0.022097086912079608f;   // 1/sqrt(2048)
#pragma unroll
        for (int i = 0; i < 4; ++i)
#pragma unroll
            for (int j = 0; j < 2; ++j) {
                const int colL   = wnw * 32 + j * 16 + c;
                const float ab   = sXi[colL];
                const float hab2 = 0.5f * ab * ab;
                const size_t col = (size_t)(n0 + colL);
#pragma unroll
                for (int e = 0; e < 8; ++e) {
                    const int rowL = wmw * 64 + i * 16 + e + 8 * half;
                    const float d  = acc[i][j][e];
                    const float v  = inv_sqrt_m * __expf(ab * d - sSq[rowL] * hab2);
                    out[(size_t)(m0 + rowL) * N + col] = (__bf16)v;
                }
            }
    } else {
        float* out = (float*)Out;
#pragma unroll
        for (int i = 0; i < 4; ++i)
#pragma unroll
            for (int j = 0; j < 2; ++j) {
                const size_t col = (size_t)(n0 + wnw * 32 + j * 16 + c);
#pragma unroll
                for (int e = 0; e < 8; ++e) {
                    const int rowL = wmw * 64 + i * 16 + e + 8 * half;
                    out[(size_t)(m0 + rowL) * N + col] = acc[i][j][e];
                }
            }
    }
}

// ---------------------------------------------------------------------------
extern "C" void kernel_launch(void* const* d_in, const int* in_sizes, int n_in,
                              void* d_out, int out_size, void* d_ws, size_t ws_size,
                              hipStream_t stream)
{
    const float* x   = (const float*)d_in[0];   // [8192,1024]
    const float* w   = (const float*)d_in[1];   // [4096,1024]
    const float* xis = (const float*)d_in[2];   // [2048]
    const float* gs  = (const float*)d_in[3];   // [2048,1024]

    char* ws = (char*)d_ws;
    __bf16* gs_b = (__bf16*)ws;  ws += (size_t)NRF    * DIM_K * 2;   //  4 MB
    __bf16* x_b  = (__bf16*)ws;  ws += (size_t)BSROWS * DIM_K * 2;   // 16 MB
    __bf16* w_b  = (__bf16*)ws;  ws += (size_t)OUTD   * DIM_K * 2;   //  8 MB
    float*  sq_x = (float*)ws;   ws += (size_t)BSROWS * 4;
    float*  sq_w = (float*)ws;   ws += (size_t)OUTD   * 4;
    __bf16* phi  = (__bf16*)ws;  ws += (size_t)BSROWS * NRF * 2;     // 32 MB
    __bf16* wei  = (__bf16*)ws;  ws += (size_t)OUTD   * NRF * 2;     // 16 MB

    // Phase 0: bf16 conversion + row sum-of-squares
    rowconv_kernel<<<BSROWS, 256, 0, stream>>>(x,  x_b,  sq_x, DIM_K);
    rowconv_kernel<<<OUTD,   256, 0, stream>>>(w,  w_b,  sq_w, DIM_K);
    rowconv_kernel<<<NRF,    256, 0, stream>>>(gs, gs_b, nullptr, DIM_K);

    // Phase 1: weights = phi(input_weights)   [OUTD x NRF]  (bf16)
    gemm_wmma_kernel<0><<<dim3(NRF / BN, OUTD / BM), 256, 0, stream>>>(
        w_b, gs_b, wei, sq_w, xis, OUTD, NRF, DIM_K);

    // Phase 2: phi_x = phi(x)                 [BSROWS x NRF] (bf16)
    gemm_wmma_kernel<0><<<dim3(NRF / BN, BSROWS / BM), 256, 0, stream>>>(
        x_b, gs_b, phi, sq_x, xis, BSROWS, NRF, DIM_K);

    // Phase 3: out = phi_x @ weights^T        [BSROWS x OUTD] (f32)
    gemm_wmma_kernel<1><<<dim3(OUTD / BN, BSROWS / BM), 256, 0, stream>>>(
        phi, wei, d_out, nullptr, nullptr, BSROWS, OUTD, NRF);
}
